// Attention_68238440399029
// MI455X (gfx1250) — matrix operand order repaired and checked by reading, not compiled, not run on hardware
//
#include <hip/hip_runtime.h>
#include <hip/hip_bf16.h>
#include <stdint.h>

#define HIDDEN 1024
#define NHEADS 16
#define DEPTH  64
#define BATCH  2
#define FQ     2048
#define TK     2048
#define MROWS  (BATCH * FQ)  // 4096

typedef __attribute__((ext_vector_type(16))) __bf16 v16bf;
typedef __attribute__((ext_vector_type(8)))  float  v8f;
typedef int v4i_vs __attribute__((vector_size(16)));  // matches builtin param type

union BF16Frag { v16bf v; uint4 q[2]; };

// ---- CDNA5 async global->LDS path (ASYNCcnt), with portable fallback ------
#if defined(__has_builtin)
#if __has_builtin(__builtin_amdgcn_global_load_async_to_lds_b128) && \
    __has_builtin(__builtin_amdgcn_s_wait_asynccnt)
#define HAVE_ASYNC_LDS 1
#endif
#endif

__device__ __forceinline__ void async_copy_b128(__bf16* lds_dst,
                                                const __bf16* gsrc) {
#ifdef HAVE_ASYNC_LDS
  __builtin_amdgcn_global_load_async_to_lds_b128(
      (__attribute__((address_space(1))) v4i_vs*)gsrc,
      (__attribute__((address_space(3))) v4i_vs*)lds_dst, 0, 0);
#else
  *(uint4*)lds_dst = *(const uint4*)gsrc;
#endif
}

__device__ __forceinline__ void async_copy_wait() {
#ifdef HAVE_ASYNC_LDS
  __builtin_amdgcn_s_wait_asynccnt(0);
#endif
}

__device__ __forceinline__ v8f wmma_bf16(v16bf a, v16bf b, v8f c) {
  // D = A(16x32 bf16) * B(32x16 bf16) + C(16x16 f32)
  return __builtin_amdgcn_wmma_f32_16x16x32_bf16(false, a, false, b, (short)0, c,
                                                 false, false);
}

__device__ __forceinline__ unsigned pk2bf(float a, float b) {
  unsigned short lo = __builtin_bit_cast(unsigned short, (__bf16)a);
  unsigned short hi = __builtin_bit_cast(unsigned short, (__bf16)b);
  return (unsigned)lo | ((unsigned)hi << 16);
}

// A fragment (16x32, 16-bit): lane = row (lane&15); lanes<16 hold K 0-7 & 16-23,
// lanes>=16 hold K 8-15 & 24-31.  Source row-major [row][k], stride in elements.
__device__ __forceinline__ v16bf load_a_frag(const __bf16* base, int stride) {
  const int lane = threadIdx.x & 31;
  const __bf16* p = base + (lane & 15) * stride + ((lane >> 4) << 3);
  BF16Frag f;
  f.q[0] = *(const uint4*)p;        // K base0 .. base0+7
  f.q[1] = *(const uint4*)(p + 16); // K base0+16 .. base0+23
  return f.v;
}

// B fragment (32x16, 16-bit): lane = col (lane&15); lanes<16 hold K 0-7 & 16-23,
// lanes>=16 hold K 8-15 & 24-31.  Source is K-transposed storage [col][k].
__device__ __forceinline__ v16bf load_bT_frag(const __bf16* base, int stride) {
  const int lane = threadIdx.x & 31;
  const __bf16* p = base + (lane & 15) * stride + ((lane >> 4) << 3);
  BF16Frag f;
  f.q[0] = *(const uint4*)p;
  f.q[1] = *(const uint4*)(p + 16);
  return f.v;
}

#define LDS_AS 40  // padded bf16 stride, conflict-free b128 fragment reads
#define LDS_BS 40

// ---------------------------------------------------------------------------
// Kernel 1: QKV projection.  [4096,1024]f32 x [1024,1024]f32 -> bf16, tiled
// 128x64 per block, 8 waves of 32x32 (2x2 WMMA tiles), K-step 32.
// z=0: Q (scaled, layout [b][n][f][d]); z=1: K ([b][n][t][d]); z=2: V^T ([b][n][d][t])
// ---------------------------------------------------------------------------
__global__ __launch_bounds__(256) void proj_qkv_kernel(
    const float* __restrict__ q_in, const float* __restrict__ s_in,
    const float* __restrict__ wq_, const float* __restrict__ wk_,
    const float* __restrict__ wv_,
    __bf16* __restrict__ qbf, __bf16* __restrict__ kbf, __bf16* __restrict__ vbf) {
  __shared__ __bf16 ldsA[128 * LDS_AS];
  __shared__ __bf16 ldsBT[64 * LDS_BS];

  const int t = threadIdx.x;
  const int which = blockIdx.z;
  const float* X = (which == 0) ? q_in : s_in;
  const float* W = (which == 0) ? wq_ : (which == 1 ? wk_ : wv_);
  const int rowBase = blockIdx.x * 128;
  const int colBase = blockIdx.y * 64;

  const int wid = t >> 5;
  const int waveM = wid >> 1, waveN = wid & 1;

  v8f acc[2][2];
#pragma unroll
  for (int i = 0; i < 2; ++i)
#pragma unroll
    for (int j = 0; j < 2; ++j)
#pragma unroll
      for (int v = 0; v < 8; ++v) acc[i][j][v] = 0.f;

  const int ar = t >> 1, ac0 = (t & 1) << 4;  // A staging: 2 threads/row, 16 f32 each
  const int bcol = t & 63;                    // B staging column

  for (int kk = 0; kk < HIDDEN; kk += 32) {
    {  // stage A tile 128x32 f32 -> bf16 LDS (row major, padded)
      const float* src = X + (size_t)(rowBase + ar) * HIDDEN + kk + ac0;
      __bf16* dst = &ldsA[ar * LDS_AS + ac0];
#pragma unroll
      for (int i = 0; i < 4; ++i) {
        float4 f = *(const float4*)(src + 4 * i);
        *(unsigned*)(dst + 4 * i)     = pk2bf(f.x, f.y);
        *(unsigned*)(dst + 4 * i + 2) = pk2bf(f.z, f.w);
      }
    }
    {  // stage B tile 32x64 f32 -> K-transposed bf16 LDS [n][k]
#pragma unroll
      for (int p = 0; p < 4; ++p) {
        int k = p * 8 + ((t >> 6) << 1);
        float w0 = W[(size_t)(kk + k) * HIDDEN + colBase + bcol];
        float w1 = W[(size_t)(kk + k + 1) * HIDDEN + colBase + bcol];
        *(unsigned*)&ldsBT[bcol * LDS_BS + k] = pk2bf(w0, w1);
      }
    }
    __syncthreads();

    v16bf aF[2], bF[2];
#pragma unroll
    for (int i = 0; i < 2; ++i)
      aF[i] = load_a_frag(&ldsA[(waveM * 32 + i * 16) * LDS_AS], LDS_AS);
#pragma unroll
    for (int j = 0; j < 2; ++j)
      bF[j] = load_bT_frag(&ldsBT[(waveN * 32 + j * 16) * LDS_BS], LDS_BS);
#pragma unroll
    for (int i = 0; i < 2; ++i)
#pragma unroll
      for (int j = 0; j < 2; ++j) acc[i][j] = wmma_bf16(aF[i], bF[j], acc[i][j]);
    __syncthreads();
  }

  // Epilogue: C layout = lane col (lane&15), rows (lane>>4)*8 + v.
  // The `which` branch is lifted OUTSIDE the loop nest so each grid.z plane
  // is a straight-line store loop (no per-element branches).
  const int lane = t & 31;
  const int cl = lane & 15;
  const int rh = (lane >> 4) << 3;
  const int baseR = rowBase + waveM * 32;
  const int baseC = colBase + waveN * 32;

  if (which == 0) {
#pragma unroll
    for (int i = 0; i < 2; ++i)
#pragma unroll
      for (int j = 0; j < 2; ++j)
#pragma unroll
        for (int v = 0; v < 8; ++v) {
          int gr = baseR + i * 16 + rh + v;
          int gc = baseC + j * 16 + cl;
          int b = gr >> 11, fr = gr & 2047;
          int hn = gc >> 6, d = gc & 63;
          size_t bh = (size_t)(b * NHEADS + hn);
          qbf[(bh * FQ + fr) * DEPTH + d] = (__bf16)(acc[i][j][v] * 0.125f);
        }
  } else if (which == 1) {
#pragma unroll
    for (int i = 0; i < 2; ++i)
#pragma unroll
      for (int j = 0; j < 2; ++j)
#pragma unroll
        for (int v = 0; v < 8; ++v) {
          int gr = baseR + i * 16 + rh + v;
          int gc = baseC + j * 16 + cl;
          int b = gr >> 11, fr = gr & 2047;
          int hn = gc >> 6, d = gc & 63;
          size_t bh = (size_t)(b * NHEADS + hn);
          kbf[(bh * TK + fr) * DEPTH + d] = (__bf16)acc[i][j][v];
        }
  } else {
#pragma unroll
    for (int i = 0; i < 2; ++i)
#pragma unroll
      for (int j = 0; j < 2; ++j)
#pragma unroll
        for (int v = 0; v < 8; ++v) {
          int gr = baseR + i * 16 + rh + v;
          int gc = baseC + j * 16 + cl;
          int b = gr >> 11, fr = gr & 2047;
          int hn = gc >> 6, d = gc & 63;
          size_t bh = (size_t)(b * NHEADS + hn);
          vbf[(bh * DEPTH + d) * TK + fr] = (__bf16)acc[i][j][v];  // V^T
        }
  }
}

// ---------------------------------------------------------------------------
// Kernel 2: flash attention.  Block = 128 query rows for one (b, head);
// 8 waves x 16 rows, Q fragments resident.  K/V^T tiles staged with
// GLOBAL_LOAD_ASYNC_TO_LDS_B128 (ASYNCcnt), S tiles via 2xK32 WMMA + bias,
// online softmax, P through wave-private LDS, 4 P*V WMMAs per 32-key step.
// ---------------------------------------------------------------------------
#define LK 72  // padded bf16 stride for 64-wide K/V^T tiles
#define LP 40

__global__ __launch_bounds__(256) void attn_kernel(
    const __bf16* __restrict__ qbf, const __bf16* __restrict__ kbf,
    const __bf16* __restrict__ vbf, const float* __restrict__ bias,
    __bf16* __restrict__ attno) {
  __shared__ __bf16 ldsK[64 * LK];
  __shared__ __bf16 ldsVT[64 * LK];
  __shared__ __bf16 ldsP[8 * 16 * LP];

  const int t = threadIdx.x;
  const int wid = t >> 5;
  const int lane = t & 31;
  const int cl = lane & 15;
  const int rh = (lane >> 4) << 3;

  const int bn = blockIdx.y;  // b*16 + head
  const int b = bn >> 4, head = bn & 15;
  const int f0 = blockIdx.x * 128 + wid * 16;

  const __bf16* qb = qbf + ((size_t)bn * FQ + f0) * DEPTH;
  v16bf aQ[2];
  aQ[0] = load_a_frag(qb, DEPTH);       // K(d) 0..31
  aQ[1] = load_a_frag(qb + 32, DEPTH);  // K(d) 32..63

  float m[8], lsum[8];
  v8f acc[4];
#pragma unroll
  for (int v = 0; v < 8; ++v) { m[v] = -1e30f; lsum[v] = 0.f; }
#pragma unroll
  for (int dt = 0; dt < 4; ++dt)
#pragma unroll
    for (int v = 0; v < 8; ++v) acc[dt][v] = 0.f;

  __bf16* wP = &ldsP[wid * 16 * LP];
  const int sr = t >> 2, sc0 = (t & 3) << 4;  // staging: 4 threads/row, 16 bf16 each

  for (int t0 = 0; t0 < TK; t0 += 64) {
    {  // stage K tile [t][d] via async DMA to LDS
      const __bf16* src = kbf + ((size_t)bn * TK + t0 + sr) * DEPTH + sc0;
      async_copy_b128(&ldsK[sr * LK + sc0], src);
      async_copy_b128(&ldsK[sr * LK + sc0 + 8], src + 8);
    }
    {  // stage V^T tile [d][t] via async DMA to LDS
      const __bf16* src = vbf + ((size_t)bn * DEPTH + sr) * TK + t0 + sc0;
      async_copy_b128(&ldsVT[sr * LK + sc0], src);
      async_copy_b128(&ldsVT[sr * LK + sc0 + 8], src + 8);
    }
    async_copy_wait();   // s_wait_asynccnt 0
    __syncthreads();

#pragma unroll
    for (int tt = 0; tt < 64; tt += 32) {
      // ---- S = Q K^T + bias : two 16x16 tiles ----
      v8f s[2];
#pragma unroll
      for (int j = 0; j < 2; ++j) {
        const float* bp =
            bias + ((size_t)b * FQ + f0 + rh) * TK + t0 + tt + j * 16 + cl;
#pragma unroll
        for (int v = 0; v < 8; ++v) s[j][v] = bp[(size_t)v * TK];
        v16bf bK0 = load_bT_frag(&ldsK[(tt + j * 16) * LK], LK);       // d 0..31
        v16bf bK1 = load_bT_frag(&ldsK[(tt + j * 16) * LK + 32], LK);  // d 32..63
        s[j] = wmma_bf16(aQ[0], bK0, s[j]);
        s[j] = wmma_bf16(aQ[1], bK1, s[j]);
      }

      // ---- online softmax over these 32 keys ----
      float rmax[8], pa[8], pb[8], rsum[8];
#pragma unroll
      for (int v = 0; v < 8; ++v) rmax[v] = fmaxf(s[0][v], s[1][v]);
#pragma unroll
      for (int off = 1; off < 16; off <<= 1)
#pragma unroll
        for (int v = 0; v < 8; ++v)
          rmax[v] = fmaxf(rmax[v], __shfl_xor(rmax[v], off));
#pragma unroll
      for (int v = 0; v < 8; ++v) {
        float nm = fmaxf(m[v], rmax[v]);
        float sc = __expf(m[v] - nm);
        pa[v] = __expf(s[0][v] - nm);
        pb[v] = __expf(s[1][v] - nm);
        rsum[v] = pa[v] + pb[v];
        m[v] = nm;
        lsum[v] *= sc;
#pragma unroll
        for (int dt = 0; dt < 4; ++dt) acc[dt][v] *= sc;
      }
#pragma unroll
      for (int off = 1; off < 16; off <<= 1)
#pragma unroll
        for (int v = 0; v < 8; ++v) rsum[v] += __shfl_xor(rsum[v], off);
#pragma unroll
      for (int v = 0; v < 8; ++v) lsum[v] += rsum[v];

      // ---- P (16x32) -> wave-private LDS -> A fragment ----
#pragma unroll
      for (int v = 0; v < 8; ++v) {
        wP[(rh + v) * LP + cl]      = (__bf16)pa[v];
        wP[(rh + v) * LP + 16 + cl] = (__bf16)pb[v];
      }
      v16bf aP = load_a_frag(wP, LP);

      // ---- acc += P * V ----
#pragma unroll
      for (int dt = 0; dt < 4; ++dt) {
        v16bf bV = load_bT_frag(&ldsVT[(dt * 16) * LK + tt], LK);
        acc[dt] = wmma_bf16(aP, bV, acc[dt]);
      }
    }
    __syncthreads();
  }

  // normalize and store attn[b][f][head][d]
#pragma unroll
  for (int v = 0; v < 8; ++v) {
    float inv = 1.f / lsum[v];
    int fr = f0 + rh + v;
    __bf16* dst = attno + ((size_t)(b * FQ + fr)) * HIDDEN + head * DEPTH;
#pragma unroll
    for (int dt = 0; dt < 4; ++dt)
      dst[dt * 16 + cl] = (__bf16)(acc[dt][v] * inv);
  }
}

// ---------------------------------------------------------------------------
// Kernel 3: output projection.  attn(bf16)[4096,1024] x wo(f32->bf16)[1024,1024]
// -> out f32 [B,F,H].  A tile staged via async DMA to LDS.
// ---------------------------------------------------------------------------
__global__ __launch_bounds__(256) void out_proj_kernel(
    const __bf16* __restrict__ attn, const float* __restrict__ wo,
    float* __restrict__ out) {
  __shared__ __bf16 ldsA[128 * LDS_AS];
  __shared__ __bf16 ldsBT[64 * LDS_BS];

  const int t = threadIdx.x;
  const int rowBase = blockIdx.x * 128;
  const int colBase = blockIdx.y * 64;
  const int wid = t >> 5;
  const int waveM = wid >> 1, waveN = wid & 1;

  v8f acc[2][2];
#pragma unroll
  for (int i = 0; i < 2; ++i)
#pragma unroll
    for (int j = 0; j < 2; ++j)
#pragma unroll
      for (int v = 0; v < 8; ++v) acc[i][j][v] = 0.f;

  const int ar = t >> 1, ac0 = (t & 1) << 4;
  const int bcol = t & 63;

  for (int kk = 0; kk < HIDDEN; kk += 32) {
    {  // stage A (already bf16): 128x32 via async DMA to LDS
      const __bf16* src = attn + (size_t)(rowBase + ar) * HIDDEN + kk + ac0;
      async_copy_b128(&ldsA[ar * LDS_AS + ac0], src);
      async_copy_b128(&ldsA[ar * LDS_AS + ac0 + 8], src + 8);
    }
    {  // stage B transposed: wo f32 -> bf16 [n][k]
#pragma unroll
      for (int p = 0; p < 4; ++p) {
        int k = p * 8 + ((t >> 6) << 1);
        float w0 = wo[(size_t)(kk + k) * HIDDEN + colBase + bcol];
        float w1 = wo[(size_t)(kk + k + 1) * HIDDEN + colBase + bcol];
        *(unsigned*)&ldsBT[bcol * LDS_BS + k] = pk2bf(w0, w1);
      }
    }
    async_copy_wait();   // s_wait_asynccnt 0
    __syncthreads();

    v16bf aF[2], bF[2];
#pragma unroll
    for (int i = 0; i < 2; ++i)
      aF[i] = load_a_frag(&ldsA[(waveM * 32 + i * 16) * LDS_AS], LDS_AS);
#pragma unroll
    for (int j = 0; j < 2; ++j)
      bF[j] = load_bT_frag(&ldsBT[(waveN * 32 + j * 16) * LDS_BS], LDS_BS);
#pragma unroll
    for (int i = 0; i < 2; ++i)
#pragma unroll
      for (int j = 0; j < 2; ++j) acc[i][j] = wmma_bf16(aF[i], bF[j], acc[i][j]);
    __syncthreads();
  }

  const int lane = t & 31;
  const int cl = lane & 15;
  const int rh = (lane >> 4) << 3;
#pragma unroll
  for (int i = 0; i < 2; ++i)
#pragma unroll
    for (int j = 0; j < 2; ++j)
#pragma unroll
      for (int v = 0; v < 8; ++v) {
        int gr = rowBase + waveM * 32 + i * 16 + rh + v;
        int gc = colBase + waveN * 32 + j * 16 + cl;
        out[(size_t)gr * HIDDEN + gc] = acc[i][j][v];
      }
}

extern "C" void kernel_launch(void* const* d_in, const int* in_sizes, int n_in,
                              void* d_out, int out_size, void* d_ws, size_t ws_size,
                              hipStream_t stream) {
  const float* q_in = (const float*)d_in[0];
  const float* s_in = (const float*)d_in[1];
  const float* bias = (const float*)d_in[2];
  const float* wq = (const float*)d_in[3];
  const float* wk = (const float*)d_in[4];
  const float* wv = (const float*)d_in[5];
  const float* wo = (const float*)d_in[6];
  float* out = (float*)d_out;

  const size_t qkvElems = (size_t)BATCH * NHEADS * FQ * DEPTH;  // 4M elems, 8MB bf16
  __bf16* qbf = (__bf16*)d_ws;
  __bf16* kbf = qbf + qkvElems;
  __bf16* vbf = kbf + qkvElems;
  __bf16* attnb = vbf + qkvElems;  // total 32 MB of workspace

  proj_qkv_kernel<<<dim3(MROWS / 128, HIDDEN / 64, 3), 256, 0, stream>>>(
      q_in, s_in, wq, wk, wv, qbf, kbf, vbf);
  attn_kernel<<<dim3(FQ / 128, BATCH * NHEADS), 256, 0, stream>>>(
      qbf, kbf, vbf, bias, attnb);
  out_proj_kernel<<<dim3(MROWS / 128, HIDDEN / 64), 256, 0, stream>>>(
      attnb, wo, out);
}
